// InEKFParallelSO2_23622320128487
// MI455X (gfx1250) — matrix operand order repaired and checked by reading, not compiled, not run on hardware
//
#include <hip/hip_runtime.h>
#include <hip/hip_bf16.h>
#include <math.h>

// Problem dims
#define B_  2
#define T_  8192
#define H_  16
#define NB_ 32
#define D_  1024
#define DH_ 128
#define HNB 512            // H*NB
#define ROWS 16384         // B*T
#define CHUNK 128          // scan chunk length
#define NCHUNK 64          // T / CHUNK

typedef __attribute__((ext_vector_type(16))) __bf16 v16bf;
typedef __attribute__((ext_vector_type(8)))  __bf16 v8bf;
typedef __attribute__((ext_vector_type(8)))  float  v8f;

static __device__ __forceinline__ v16bf cat16(v8bf lo, v8bf hi) {
  return __builtin_shufflevector(lo, hi, 0,1,2,3,4,5,6,7,8,9,10,11,12,13,14,15);
}

// LDS byte offset of a __shared__ object (generic -> AS(3) addrspacecast).
typedef __attribute__((address_space(3))) __bf16 as3_bf16;
static __device__ __forceinline__ unsigned lds_off_u32(const __bf16* p) {
  return (unsigned)(uintptr_t)(as3_bf16*)p;
}

// CDNA5 async global->LDS b128 copy (GVS mode: saddr64 + vaddr32 offset).
static __device__ __forceinline__ void async_copy_b128(unsigned lds_off, unsigned goff,
                                                       unsigned long long sbase) {
  asm volatile("global_load_async_to_lds_b128 %0, %1, %2"
               :: "v"(lds_off), "v"(goff), "s"(sbase)
               : "memory");
}

// ---------------------------------------------------------------------------
// Kernel 1: transpose+convert W1 -> W1T(bf16, 128x1024), W2 -> W2T(bf16, 512x128),
//           compute K* (closed-form DARE) -> Kst/Alpha workspace + third output.
// ---------------------------------------------------------------------------
__global__ void prep_kernel(const float* __restrict__ W1, const float* __restrict__ W2,
                            const float* __restrict__ logQ, const float* __restrict__ logR,
                            __bf16* __restrict__ W1T, __bf16* __restrict__ W2T,
                            float* __restrict__ Kst, float* __restrict__ Alp,
                            float* __restrict__ outK) {
  int idx = blockIdx.x * blockDim.x + threadIdx.x;
  if (idx < DH_ * D_) {                       // W1T[n][k] = W1[k][n]
    int n = idx >> 10, k = idx & (D_ - 1);
    W1T[idx] = (__bf16)W1[k * DH_ + n];
  } else if (idx < DH_ * D_ + HNB * DH_) {    // W2T[n][k] = W2[k][n]
    int j = idx - DH_ * D_;
    int n = j >> 7, k = j & (DH_ - 1);
    W2T[j] = (__bf16)W2[k * HNB + n];
  } else if (idx < DH_ * D_ + HNB * DH_ + HNB) {
    int j = idx - (DH_ * D_ + HNB * DH_);
    float Q = expf(logQ[j]), R = expf(logR[j]);
    float Ppost = 0.5f * (-Q + sqrtf(Q * Q + 4.0f * Q * R));
    float Ppred = Ppost + Q;
    float K = Ppred / (Ppred + R);
    Kst[j] = K;
    Alp[j] = 1.0f - K;
    outK[j] = K;
  }
}

// ---------------------------------------------------------------------------
// Kernel 2: fused MLP (bf16 WMMA) + pi*tanh + angle wrap -> nu (f32, ROWS x 512)
// Block = 256 threads = 8 waves; each wave owns a 16-row slab.
// W1T k-slices (128x32 bf16 = 8KB) are async-DMA'd into double-buffered LDS
// once per block and shared by all 8 waves.
// ---------------------------------------------------------------------------
#define BPAD 40   // 32 bf16 + 8 pad -> 20-dword row stride, conflict-free halves

__global__ void __launch_bounds__(256)
mlp_nu_kernel(const float* __restrict__ X,      // content_emb (ROWS x D)
              const float* __restrict__ b1,
              const float* __restrict__ b2,
              const float* __restrict__ theta,  // (ROWS x 512)
              const __bf16* __restrict__ W1T,   // (128 x 1024)
              const __bf16* __restrict__ W2T,   // (512 x 128)
              float* __restrict__ nu) {         // (ROWS x 512)
  __shared__ __bf16 hidT[8][16][DH_ + 8];              // per-wave hid tile, padded
  __shared__ alignas(16) __bf16 Bs[2][DH_][BPAD];      // double-buffered W1T slice

  const int lane = threadIdx.x & 31;
  const int wave = threadIdx.x >> 5;
  const int slab = blockIdx.x * 8 + wave;
  const int rowbase = slab * 16;
  const int half = lane >> 4;                   // 0 | 1
  const int l16  = lane & 15;

  const unsigned long long w1t_base = (unsigned long long)(uintptr_t)W1T;
  const int tid = threadIdx.x;

  // Cooperative async stage of one 128x32 bf16 slice: 512 b128 transfers, 2/thread.
  auto issue_slice = [&](int kb, int buf) {
    #pragma unroll
    for (int rep = 0; rep < 2; ++rep) {
      int i   = tid + rep * 256;
      int row = i >> 2;                         // 0..127
      int seg = i & 3;                          // 4 x 8 bf16 = 32 bf16 per row
      unsigned goff = (unsigned)(row * (D_ * 2) + kb * 2 + seg * 16);
      unsigned loff = lds_off_u32(&Bs[buf][row][seg * 8]);
      async_copy_b128(loff, goff, w1t_base);
    }
  };

  // ---------------- GEMM1: hid = gelu(X @ W1 + b1), 16x128 tile ----------------
  v8f acc[8];
  #pragma unroll
  for (int i = 0; i < 8; ++i) acc[i] = (v8f){0.f,0.f,0.f,0.f,0.f,0.f,0.f,0.f};

  const float* Arow = X + (size_t)(rowbase + l16) * D_;

  issue_slice(0, 0);                            // prologue: slice 0 in flight

  #pragma unroll 1
  for (int ks = 0; ks < D_ / 32; ++ks) {
    const int kb  = ks * 32;
    const int cur = ks & 1;
    const bool has_next = (ks + 1) < (D_ / 32);
    if (has_next) issue_slice(kb + 32, cur ^ 1);

    // A fragment while DMA completes: lane l16 = row M; halves split K runs.
    __builtin_prefetch(Arow + kb + 128, 0, 1);
    v16bf afrag;
    {
      const v8f f0 = *(const v8f*)(Arow + kb + half * 8);
      const v8f f1 = *(const v8f*)(Arow + kb + half * 8 + 16);
      #pragma unroll
      for (int i = 0; i < 8; ++i) { afrag[i] = (__bf16)f0[i]; afrag[8 + i] = (__bf16)f1[i]; }
    }

    if (has_next) asm volatile("s_wait_asynccnt 2" ::: "memory");
    else          asm volatile("s_wait_asynccnt 0" ::: "memory");
    __syncthreads();                            // producer -> consumers

    #pragma unroll
    for (int nt = 0; nt < 8; ++nt) {
      const __bf16* bp = &Bs[cur][nt * 16 + l16][half * 8];
      v16bf bfrag = cat16(*(const v8bf*)bp, *(const v8bf*)(bp + 16));
      acc[nt] = __builtin_amdgcn_wmma_f32_16x16x32_bf16(
          false, afrag, false, bfrag, (short)0, acc[nt], false, false);
    }
    __syncthreads();                            // consumers done -> next overwrite ok
  }

  // bias + exact GELU, stash bf16 tile to LDS in (row, col) order
  #pragma unroll
  for (int nt = 0; nt < 8; ++nt) {
    const float bias = b1[nt * 16 + l16];
    #pragma unroll
    for (int r = 0; r < 8; ++r) {
      float x = acc[nt][r] + bias;
      float g = 0.5f * x * (1.0f + erff(x * 0.70710678118654752f));
      hidT[wave][r + half * 8][nt * 16 + l16] = (__bf16)g;
    }
  }
  __syncthreads();

  // ---------------- GEMM2: z = hid @ W2 + b2, then wrap ----------------
  v16bf ha[4];                                  // 4 K-steps of 32 (K=128 total)
  #pragma unroll
  for (int kk = 0; kk < 4; ++kk) {
    const __bf16* p = &hidT[wave][l16][kk * 32 + half * 8];
    ha[kk] = cat16(*(const v8bf*)p, *(const v8bf*)(p + 16));
  }

  #pragma unroll 1
  for (int nt2 = 0; nt2 < 32; ++nt2) {
    v8f c2 = (v8f){0.f,0.f,0.f,0.f,0.f,0.f,0.f,0.f};
    #pragma unroll
    for (int kk = 0; kk < 4; ++kk) {
      const __bf16* bp = W2T + (size_t)(nt2 * 16 + l16) * DH_ + kk * 32 + half * 8;
      v16bf bfrag = cat16(*(const v8bf*)bp, *(const v8bf*)(bp + 16));
      c2 = __builtin_amdgcn_wmma_f32_16x16x32_bf16(
          false, ha[kk], false, bfrag, (short)0, c2, false, false);
    }
    const int col = nt2 * 16 + l16;
    const float bias2 = b2[col];
    #pragma unroll
    for (int r = 0; r < 8; ++r) {
      const int row = rowbase + r + half * 8;
      float z = 3.14159265358979323f * tanhf(c2[r] + bias2);
      float dphi = z - theta[(size_t)row * HNB + col];
      nu[(size_t)row * HNB + col] = atan2f(sinf(dphi), cosf(dphi));
    }
  }
}

// ---------------------------------------------------------------------------
// Kernel 3: local scan within chunks of 128. d_t = a*d_{t-1} + K*nu_t.
// ---------------------------------------------------------------------------
__global__ void __launch_bounds__(512)
scan_local_kernel(const float* __restrict__ nu, const float* __restrict__ Kst,
                  const float* __restrict__ Alp, float* __restrict__ dlocal,
                  float* __restrict__ chunkend) {
  const int ch = threadIdx.x;
  const int blk = blockIdx.x;
  const int b = blk >> 6, c = blk & (NCHUNK - 1);
  const float K = Kst[ch], a = Alp[ch];
  size_t base = ((size_t)b * T_ + (size_t)c * CHUNK) * HNB + ch;
  float d = 0.f;
  #pragma unroll 4
  for (int i = 0; i < CHUNK; ++i) {
    float v = nu[base + (size_t)i * HNB];
    d = a * d + K * v;
    dlocal[base + (size_t)i * HNB] = d;
  }
  chunkend[(size_t)blk * HNB + ch] = d;
}

// ---------------------------------------------------------------------------
// Kernel 4: sequential carry scan over 64 chunk sums per channel.
// ---------------------------------------------------------------------------
__global__ void __launch_bounds__(512)
scan_carry_kernel(const float* __restrict__ chunkend, const float* __restrict__ Alp,
                  float* __restrict__ carryExcl) {
  const int ch = threadIdx.x;
  const int b = blockIdx.x;
  const float a = Alp[ch];
  float aL = a;
  #pragma unroll
  for (int j = 0; j < 7; ++j) aL *= aL;         // a^128
  float carry = 0.f;
  for (int c = 0; c < NCHUNK; ++c) {
    size_t idx = (size_t)(b * NCHUNK + c) * HNB + ch;
    carryExcl[idx] = carry;
    carry = chunkend[idx] + aL * carry;
  }
}

// ---------------------------------------------------------------------------
// Kernel 5: fixup d_t = dlocal_t + a^(i+1)*carry, emit d and theta_hat.
// ---------------------------------------------------------------------------
__global__ void __launch_bounds__(512)
scan_fix_kernel(const float* __restrict__ theta, const float* __restrict__ Alp,
                const float* __restrict__ carryExcl, float* __restrict__ out_theta,
                float* __restrict__ out_d) {
  const int ch = threadIdx.x;
  const int blk = blockIdx.x;
  const int b = blk >> 6, c = blk & (NCHUNK - 1);
  const float a = Alp[ch];
  const float carry = carryExcl[(size_t)blk * HNB + ch];
  size_t base = ((size_t)b * T_ + (size_t)c * CHUNK) * HNB + ch;
  float w = a;
  #pragma unroll 4
  for (int i = 0; i < CHUNK; ++i) {
    size_t idx = base + (size_t)i * HNB;
    float dv = out_d[idx] + w * carry;
    w *= a;
    out_d[idx] = dv;
    out_theta[idx] = theta[idx] + dv;
  }
}

// ---------------------------------------------------------------------------
extern "C" void kernel_launch(void* const* d_in, const int* in_sizes, int n_in,
                              void* d_out, int out_size, void* d_ws, size_t ws_size,
                              hipStream_t stream) {
  const float* theta = (const float*)d_in[0];   // (B,T,H,NB)
  const float* X     = (const float*)d_in[1];   // (B,T,D)
  const float* W1    = (const float*)d_in[2];   // (D,DH)
  const float* b1    = (const float*)d_in[3];   // (DH)
  const float* W2    = (const float*)d_in[4];   // (DH,H*NB)
  const float* b2    = (const float*)d_in[5];   // (H*NB)
  const float* logQ  = (const float*)d_in[6];   // (H,NB)
  const float* logR  = (const float*)d_in[7];   // (H,NB)

  const size_t NEL = (size_t)ROWS * HNB;        // 8388608
  float* out_theta = (float*)d_out;
  float* out_d     = out_theta + NEL;
  float* out_K     = out_d + NEL;

  // Workspace carve-up (bytes)
  char* ws = (char*)d_ws;
  __bf16* W1T      = (__bf16*)(ws + 0);                      // 262144 B
  __bf16* W2T      = (__bf16*)(ws + 262144);                 // 131072 B
  float*  Kst      = (float*) (ws + 393216);                 //   2048 B
  float*  Alp      = (float*) (ws + 395264);                 //   2048 B
  float*  chunkend = (float*) (ws + 397312);                 // 262144 B
  float*  carryEx  = (float*) (ws + 659456);                 // 262144 B
  float*  nu       = (float*) (ws + 921600);                 // 33554432 B

  // 1) weight transpose/convert + K*
  {
    int n = DH_ * D_ + HNB * DH_ + HNB;
    prep_kernel<<<(n + 255) / 256, 256, 0, stream>>>(W1, W2, logQ, logR,
                                                     W1T, W2T, Kst, Alp, out_K);
  }
  // 2) fused bf16-WMMA MLP (async-LDS staged weights) + wrap -> nu
  mlp_nu_kernel<<<ROWS / (16 * 8), 256, 0, stream>>>(X, b1, b2, theta, W1T, W2T, nu);
  // 3) chunked linear-recurrence scan
  scan_local_kernel<<<B_ * NCHUNK, HNB, 0, stream>>>(nu, Kst, Alp, out_d, chunkend);
  scan_carry_kernel<<<B_, HNB, 0, stream>>>(chunkend, Alp, carryEx);
  scan_fix_kernel<<<B_ * NCHUNK, HNB, 0, stream>>>(theta, Alp, carryEx, out_theta, out_d);
}
